// VAE_77833397338467
// MI455X (gfx1250) — matrix operand order repaired and checked by reading, not compiled, not run on hardware
//
#include <hip/hip_runtime.h>

// ---------------------------------------------------------------------------
// Dims from the reference
#define BN   128   // batch
#define TN   256   // time steps
#define INN  128   // IN
#define GG   5     // genre
#define HN   512   // hidden
#define KIN  160   // IN+G=133 padded to multiple of 32
#define G3   1536  // 3*H
#define LL   256   // L
#define EE   1024  // 2*H
#define LG   251   // L - G
#define HSTR 520   // LDS row stride for encoder h (pad to dodge bank conflicts)
#define XB_BYTES (BN * KIN * 2)   // one staged x timestep in LDS (40 KB)

typedef __attribute__((ext_vector_type(16))) __bf16 bf16x16;
typedef __attribute__((ext_vector_type(16))) unsigned short ushortx16;
typedef __attribute__((ext_vector_type(8)))  float   floatx8;

__device__ __forceinline__ unsigned short f2bf(float f) {
    unsigned int u = __float_as_uint(f);
    unsigned int r = u + 0x7FFFu + ((u >> 16) & 1u);   // round-to-nearest-even
    return (unsigned short)(r >> 16);
}
__device__ __forceinline__ float bf2f(unsigned short b) {
    return __uint_as_float(((unsigned int)b) << 16);
}
__device__ __forceinline__ float sigmoidf(float x) {
    return 1.0f / (1.0f + __expf(-x));
}

__device__ __forceinline__ floatx8 wmma_bf16(bf16x16 a, bf16x16 b, floatx8 c) {
    // D = A(16x32) * B(32x16) + C, f32 accumulate
    return __builtin_amdgcn_wmma_f32_16x16x32_bf16(false, a, false, b,
                                                   (short)0, c, false, false);
}

// ---- CDNA5 async global->LDS copy (ASYNCcnt-tracked) ----------------------
__device__ __forceinline__ void async_b128_to_lds(unsigned lds_byte, const void* g) {
    asm volatile("global_load_async_to_lds_b128 %0, %1, off"
                 :: "v"(lds_byte), "v"(g) : "memory");
}
__device__ __forceinline__ void wait_async() {
    asm volatile("s_wait_asynccnt 0x0" ::: "memory");
}
// Stage one x timestep (BN*KIN bf16 = 40 KB) into LDS, spread over 1024 threads.
__device__ __forceinline__ void stage_x(const unsigned short* __restrict__ Xt,
                                        unsigned lds_base_byte, int tid) {
#pragma unroll 1
    for (int i = tid * 16; i < XB_BYTES; i += 1024 * 16)
        async_b128_to_lds(lds_base_byte + (unsigned)i, (const char*)Xt + i);
}

// A fragment (16x32 bf16), row-major source with leading dim ld.
// lane l: row m0+(l&15); K elems [kb..kb+7] and [16+kb..16+kb+7], kb=8*(l>>4).
__device__ __forceinline__ bf16x16 load_a_frag(const unsigned short* base, int ld,
                                               int m0, int k0, int lane) {
    const int m  = m0 + (lane & 15);
    const int kb = k0 + ((lane >> 4) << 3);
    const unsigned short* p = base + (size_t)m * ld + kb;
    ushortx16 v;
#pragma unroll
    for (int i = 0; i < 8; ++i) { v[i] = p[i]; v[8 + i] = p[16 + i]; }
    union { ushortx16 u; bf16x16 b; } cv; cv.u = v;
    return cv.b;
}

// B fragment (32x16): B[k][n] = W[n][k], W row-major (rows of length ld).
// lane l: weight row n0+(l&15); K elems [kb..kb+7] and [16+kb..16+kb+7], kb=8*(l>>4).
__device__ __forceinline__ bf16x16 load_b_frag(const unsigned short* W, int ld,
                                               int n0, int k0, int lane) {
    const int n  = n0 + (lane & 15);
    const int kb = k0 + ((lane >> 4) << 3);
    const unsigned short* p = W + (size_t)n * ld + kb;
    ushortx16 v;
#pragma unroll
    for (int i = 0; i < 8; ++i) { v[i] = p[i]; v[8 + i] = p[16 + i]; }
    union { ushortx16 u; bf16x16 b; } cv; cv.u = v;
    return cv.b;
}

// ---------------------------------------------------------------------------
// f32 -> bf16 with K padding (zeros beyond sc)
__global__ void cvt_pad_kernel(const float* __restrict__ src, unsigned short* __restrict__ dst,
                               int rows, int sc, int dc) {
    int idx = blockIdx.x * blockDim.x + threadIdx.x;
    if (idx >= rows * dc) return;
    int r = idx / dc, c = idx - r * dc;
    dst[idx] = (c < sc) ? f2bf(src[(size_t)r * sc + c]) : (unsigned short)0;
}

// x (B,T,133) f32 -> Xbf (T,B,160) bf16
__global__ void xpad_kernel(const float* __restrict__ x, unsigned short* __restrict__ dst) {
    int idx = blockIdx.x * blockDim.x + threadIdx.x;
    if (idx >= TN * BN * KIN) return;
    int k  = idx % KIN;
    int bt = idx / KIN;
    int b  = bt % BN;
    int t  = bt / BN;
    dst[idx] = (k < INN + GG) ? f2bf(x[((size_t)b * TN + t) * (INN + GG) + k])
                              : (unsigned short)0;
}

// ---------------------------------------------------------------------------
// Bidirectional encoder GRU: one persistent workgroup per direction.
// h kept in LDS as bf16; x timestep tiles double-buffered in LDS via
// global_load_async_to_lds_b128; gates via v_wmma_f32_16x16x32_bf16.
__global__ __launch_bounds__(1024) void gru_encoder_kernel(
    const unsigned short* __restrict__ Xbf,
    const unsigned short* __restrict__ Wih0, const unsigned short* __restrict__ Whh0,
    const float* __restrict__ bih0, const float* __restrict__ bhh0,
    const unsigned short* __restrict__ Wih1, const unsigned short* __restrict__ Whh1,
    const float* __restrict__ bih1, const float* __restrict__ bhh1,
    float* __restrict__ hidden /* (128,1024): [:, :512]=fwd, [:, 512:]=bwd */) {
    extern __shared__ unsigned short sm[];
    unsigned short* hl = sm;                       // [BN][HSTR] bf16
    unsigned short* xb = sm + BN * HSTR;           // 2 x [BN][KIN] bf16
    const unsigned xb_base =
        (unsigned)__builtin_amdgcn_groupstaticsize() + (unsigned)(BN * HSTR * 2);

    const int dir = blockIdx.x;
    const unsigned short* Wih = dir ? Wih1 : Wih0;
    const unsigned short* Whh = dir ? Whh1 : Whh0;
    const float* bih = dir ? bih1 : bih0;
    const float* bhh = dir ? bhh1 : bhh0;

    const int tid  = threadIdx.x;
    const int lane = tid & 31;
    const int wave = tid >> 5;
    const int tm   = (wave >> 2) * 16;   // 8 tile-rows, 4 waves per row
    const int tc0  = (wave & 3) * 8;     // 8 column tiles per wave

    // stage x(t=first) into buffer 0 while zeroing h
    {
        const int t0 = dir ? (TN - 1) : 0;
        stage_x(Xbf + (size_t)t0 * BN * KIN, xb_base, tid);
    }
    for (int i = tid; i < BN * HSTR; i += 1024) hl[i] = 0;
    wait_async();
    __syncthreads();

    float newh[8][8];
    for (int s = 0; s < TN; ++s) {
        const unsigned short* xcur = xb + (s & 1) * BN * KIN;
        if (s + 1 < TN) {   // async prefetch of next timestep into the other buffer
            const int tn_ = dir ? (TN - 2 - s) : (s + 1);
            stage_x(Xbf + (size_t)tn_ * BN * KIN,
                    xb_base + (unsigned)(((s + 1) & 1) * XB_BYTES), tid);
        }
#pragma unroll 1
        for (int ct = 0; ct < 8; ++ct) {
            const int n0 = (tc0 + ct) * 16;
            floatx8 aR = {}, aZ = {}, aIN = {}, aHN = {};
#pragma unroll 2
            for (int k0 = 0; k0 < HN; k0 += 32) {     // recurrent part (LDS h)
                bf16x16 a  = load_a_frag(hl, HSTR, tm, k0, lane);
                bf16x16 br = load_b_frag(Whh, HN, n0,            k0, lane);
                bf16x16 bz = load_b_frag(Whh, HN, HN + n0,       k0, lane);
                bf16x16 bn = load_b_frag(Whh, HN, 2 * HN + n0,   k0, lane);
                aR  = wmma_bf16(a, br, aR);
                aZ  = wmma_bf16(a, bz, aZ);
                aHN = wmma_bf16(a, bn, aHN);
            }
#pragma unroll
            for (int k0 = 0; k0 < KIN; k0 += 32) {    // input part (LDS-staged x)
                bf16x16 a  = load_a_frag(xcur, KIN, tm, k0, lane);
                bf16x16 br = load_b_frag(Wih, KIN, n0,          k0, lane);
                bf16x16 bz = load_b_frag(Wih, KIN, HN + n0,     k0, lane);
                bf16x16 bn = load_b_frag(Wih, KIN, 2 * HN + n0, k0, lane);
                aR  = wmma_bf16(a, br, aR);
                aZ  = wmma_bf16(a, bz, aZ);
                aIN = wmma_bf16(a, bn, aIN);
            }
            const int n = n0 + (lane & 15);
            const float br_  = bih[n] + bhh[n];
            const float bz_  = bih[HN + n] + bhh[HN + n];
            const float bin_ = bih[2 * HN + n];
            const float bhn_ = bhh[2 * HN + n];
#pragma unroll
            for (int r = 0; r < 8; ++r) {
                const int m  = tm + r + ((lane >> 4) << 3);
                const float hold = bf2f(hl[m * HSTR + n]);
                const float rg = sigmoidf(aR[r] + br_);
                const float zg = sigmoidf(aZ[r] + bz_);
                const float ng = tanhf(aIN[r] + bin_ + rg * (aHN[r] + bhn_));
                newh[ct][r] = (1.0f - zg) * ng + zg * hold;
            }
        }
        __syncthreads();
#pragma unroll 1
        for (int ct = 0; ct < 8; ++ct) {
            const int n = (tc0 + ct) * 16 + (lane & 15);
#pragma unroll
            for (int r = 0; r < 8; ++r) {
                const int m = tm + r + ((lane >> 4) << 3);
                hl[m * HSTR + n] = f2bf(newh[ct][r]);
            }
        }
        wait_async();     // next x buffer fully staged (copy hid behind compute)
        __syncthreads();
    }
    float* hout = hidden + (dir ? HN : 0);
    for (int ct = 0; ct < 8; ++ct) {
        const int n = (tc0 + ct) * 16 + (lane & 15);
        for (int r = 0; r < 8; ++r) {
            const int m = tm + r + ((lane >> 4) << 3);
            hout[(size_t)m * EE + n] = newh[ct][r];
        }
    }
}

// ---------------------------------------------------------------------------
// Decoder GRU with output feedback: one persistent workgroup.
__global__ __launch_bounds__(1024) void gru_decoder_kernel(
    const float* __restrict__ x,            // (B,T,133) for note0
    const float* __restrict__ dec_h,        // (B,H)
    const unsigned short* __restrict__ Wih, // (1536,128) bf16
    const unsigned short* __restrict__ Whh, // (1536,512) bf16
    const float* __restrict__ bih, const float* __restrict__ bhh,
    const unsigned short* __restrict__ Wd,  // (128,512) bf16  W_dec1
    const float* __restrict__ bd,
    float* __restrict__ out /* (B,T,IN) */) {
    extern __shared__ unsigned short sm[];
    unsigned short* hl = sm;                 // [BN][HN]  bf16
    unsigned short* nl = sm + BN * HN;       // [BN][INN] bf16

    const int tid  = threadIdx.x;
    const int lane = tid & 31;
    const int wave = tid >> 5;
    const int tm   = (wave >> 2) * 16;
    const int tc0  = (wave & 3) * 8;

    for (int i = tid; i < BN * HN; i += 1024) hl[i] = f2bf(dec_h[i]);
    for (int i = tid; i < BN * INN; i += 1024) {
        const int b = i >> 7, k = i & 127;
        const float v = x[(size_t)b * TN * (INN + GG) + k];   // x[b][0][k]
        nl[i] = f2bf(v);
        out[((size_t)b * TN) * INN + k] = v;                  // output[:,0,:]
    }
    __syncthreads();

    float newh[8][8];
    for (int s = 1; s < TN; ++s) {
#pragma unroll 1
        for (int ct = 0; ct < 8; ++ct) {
            const int n0 = (tc0 + ct) * 16;
            floatx8 aR = {}, aZ = {}, aIN = {}, aHN = {};
#pragma unroll 2
            for (int k0 = 0; k0 < HN; k0 += 32) {
                bf16x16 a  = load_a_frag(hl, HN, tm, k0, lane);
                bf16x16 br = load_b_frag(Whh, HN, n0,          k0, lane);
                bf16x16 bz = load_b_frag(Whh, HN, HN + n0,     k0, lane);
                bf16x16 bn = load_b_frag(Whh, HN, 2 * HN + n0, k0, lane);
                aR  = wmma_bf16(a, br, aR);
                aZ  = wmma_bf16(a, bz, aZ);
                aHN = wmma_bf16(a, bn, aHN);
            }
#pragma unroll
            for (int k0 = 0; k0 < INN; k0 += 32) {
                bf16x16 a  = load_a_frag(nl, INN, tm, k0, lane);
                bf16x16 br = load_b_frag(Wih, INN, n0,          k0, lane);
                bf16x16 bz = load_b_frag(Wih, INN, HN + n0,     k0, lane);
                bf16x16 bn = load_b_frag(Wih, INN, 2 * HN + n0, k0, lane);
                aR  = wmma_bf16(a, br, aR);
                aZ  = wmma_bf16(a, bz, aZ);
                aIN = wmma_bf16(a, bn, aIN);
            }
            const int n = n0 + (lane & 15);
            const float br_  = bih[n] + bhh[n];
            const float bz_  = bih[HN + n] + bhh[HN + n];
            const float bin_ = bih[2 * HN + n];
            const float bhn_ = bhh[2 * HN + n];
#pragma unroll
            for (int r = 0; r < 8; ++r) {
                const int m  = tm + r + ((lane >> 4) << 3);
                const float hold = bf2f(hl[m * HN + n]);
                const float rg = sigmoidf(aR[r] + br_);
                const float zg = sigmoidf(aZ[r] + bz_);
                const float ng = tanhf(aIN[r] + bin_ + rg * (aHN[r] + bhn_));
                newh[ct][r] = (1.0f - zg) * ng + zg * hold;
            }
        }
        __syncthreads();
#pragma unroll 1
        for (int ct = 0; ct < 8; ++ct) {
            const int n = (tc0 + ct) * 16 + (lane & 15);
#pragma unroll
            for (int r = 0; r < 8; ++r) {
                const int m = tm + r + ((lane >> 4) << 3);
                hl[m * HN + n] = f2bf(newh[ct][r]);
            }
        }
        __syncthreads();
        // note = h_new @ W_dec1.T + b_dec1  (128x128 out = 64 tiles, 2 per wave)
#pragma unroll 1
        for (int j = 0; j < 2; ++j) {
            const int n0 = ((wave & 3) * 2 + j) * 16;
            floatx8 acc = {};
#pragma unroll 2
            for (int k0 = 0; k0 < HN; k0 += 32) {
                bf16x16 a = load_a_frag(hl, HN, tm, k0, lane);
                bf16x16 b = load_b_frag(Wd, HN, n0, k0, lane);
                acc = wmma_bf16(a, b, acc);
            }
            const int n  = n0 + (lane & 15);
            const float bb = bd[n];
#pragma unroll
            for (int r = 0; r < 8; ++r) {
                const int m = tm + r + ((lane >> 4) << 3);
                const float v = acc[r] + bb;
                out[((size_t)m * TN + s) * INN + n] = v;
                nl[m * INN + n] = f2bf(v);
            }
        }
        __syncthreads();
    }
}

// ---------------------------------------------------------------------------
// Small f32 FC (the MLPs are <0.5% of total FLOPs)
__global__ void fc_kernel(const float* __restrict__ A, const float* __restrict__ W,
                          const float* __restrict__ bias, float* __restrict__ out,
                          int M, int N, int K, int leaky) {
    int idx = blockIdx.x * blockDim.x + threadIdx.x;
    if (idx >= M * N) return;
    int m = idx / N, n = idx - m * N;
    const float* a = A + (size_t)m * K;
    const float* w = W + (size_t)n * K;
    float s = bias[n];
    for (int k = 0; k < K; ++k) s = fmaf(a[k], w[k], s);
    out[idx] = leaky ? (s >= 0.0f ? s : 0.01f * s) : s;
}

// z = mu + exp(0.5*logvar)*eps ; concat genre -> zc (B, L)
__global__ void zcat_kernel(const float* __restrict__ mu, const float* __restrict__ lv,
                            const float* __restrict__ eps, const float* __restrict__ genre,
                            float* __restrict__ zc) {
    int idx = blockIdx.x * blockDim.x + threadIdx.x;
    if (idx >= BN * LL) return;
    int b = idx >> 8, j = idx & 255;
    float v;
    if (j < LG) v = mu[b * LG + j] + __expf(0.5f * lv[b * LG + j]) * eps[b * LG + j];
    else        v = genre[b * GG + (j - LG)];
    zc[idx] = v;
}

// ---------------------------------------------------------------------------
extern "C" void kernel_launch(void* const* d_in, const int* in_sizes, int n_in,
                              void* d_out, int out_size, void* d_ws, size_t ws_size,
                              hipStream_t stream) {
    const float* x      = (const float*)d_in[0];
    const float* genre  = (const float*)d_in[1];
    const float* eps    = (const float*)d_in[2];
    const float* Wih_e  = (const float*)d_in[3];
    const float* Whh_e  = (const float*)d_in[4];
    const float* bih_e  = (const float*)d_in[5];
    const float* bhh_e  = (const float*)d_in[6];
    const float* Wih_be = (const float*)d_in[7];
    const float* Whh_be = (const float*)d_in[8];
    const float* bih_be = (const float*)d_in[9];
    const float* bhh_be = (const float*)d_in[10];
    const float* Wih_d  = (const float*)d_in[11];
    const float* Whh_d  = (const float*)d_in[12];
    const float* bih_d  = (const float*)d_in[13];
    const float* bhh_d  = (const float*)d_in[14];
    const float* W_mu0  = (const float*)d_in[15];
    const float* b_mu0  = (const float*)d_in[16];
    const float* W_mu1  = (const float*)d_in[17];
    const float* b_mu1  = (const float*)d_in[18];
    const float* W_lv0  = (const float*)d_in[19];
    const float* b_lv0  = (const float*)d_in[20];
    const float* W_lv1  = (const float*)d_in[21];
    const float* b_lv1  = (const float*)d_in[22];
    const float* W_lat0 = (const float*)d_in[23];
    const float* b_lat0 = (const float*)d_in[24];
    const float* W_lat1 = (const float*)d_in[25];
    const float* b_lat1 = (const float*)d_in[26];
    const float* W_dec1 = (const float*)d_in[27];
    const float* b_dec1 = (const float*)d_in[28];

    // workspace carve (256B aligned)
    char* wp = (char*)d_ws;
    auto carve = [&](size_t bytes) -> void* {
        void* p = (void*)wp;
        wp += (bytes + 255) & ~(size_t)255;
        return p;
    };
    unsigned short* Xbf    = (unsigned short*)carve((size_t)TN * BN * KIN * 2);
    unsigned short* WihE   = (unsigned short*)carve((size_t)G3 * KIN * 2);
    unsigned short* WhhE   = (unsigned short*)carve((size_t)G3 * HN * 2);
    unsigned short* WihBE  = (unsigned short*)carve((size_t)G3 * KIN * 2);
    unsigned short* WhhBE  = (unsigned short*)carve((size_t)G3 * HN * 2);
    unsigned short* WihD   = (unsigned short*)carve((size_t)G3 * INN * 2);
    unsigned short* WhhD   = (unsigned short*)carve((size_t)G3 * HN * 2);
    unsigned short* WdB    = (unsigned short*)carve((size_t)INN * HN * 2);
    float* hidden = (float*)carve((size_t)BN * EE * 4);
    float* t_mu   = (float*)carve((size_t)BN * EE * 4);
    float* t_lv   = (float*)carve((size_t)BN * EE * 4);
    float* zc     = (float*)carve((size_t)BN * LL * 4);
    float* t_lat  = (float*)carve((size_t)BN * LL * 4);
    float* dech   = (float*)carve((size_t)BN * HN * 4);

    float* out_notes = (float*)d_out;                       // (B,T,IN)
    float* out_mu    = out_notes + (size_t)BN * TN * INN;   // (B,251)
    float* out_lv    = out_mu + (size_t)BN * LG;            // (B,251)

    auto g1 = [](int n) { return dim3((unsigned)((n + 255) / 256)); };

    // 1) bf16 packing
    cvt_pad_kernel<<<g1(G3 * KIN), 256, 0, stream>>>(Wih_e,  WihE,  G3, INN + GG, KIN);
    cvt_pad_kernel<<<g1(G3 * KIN), 256, 0, stream>>>(Wih_be, WihBE, G3, INN + GG, KIN);
    cvt_pad_kernel<<<g1(G3 * HN),  256, 0, stream>>>(Whh_e,  WhhE,  G3, HN, HN);
    cvt_pad_kernel<<<g1(G3 * HN),  256, 0, stream>>>(Whh_be, WhhBE, G3, HN, HN);
    cvt_pad_kernel<<<g1(G3 * INN), 256, 0, stream>>>(Wih_d,  WihD,  G3, INN, INN);
    cvt_pad_kernel<<<g1(G3 * HN),  256, 0, stream>>>(Whh_d,  WhhD,  G3, HN, HN);
    cvt_pad_kernel<<<g1(INN * HN), 256, 0, stream>>>(W_dec1, WdB,   INN, HN, HN);
    xpad_kernel<<<g1(TN * BN * KIN), 256, 0, stream>>>(x, Xbf);

    // 2) bidirectional encoder (persistent WMMA recurrence; 1 WGP per direction;
    //    x timesteps double-buffered in LDS via async global->LDS copies)
    const size_t enc_lds = (size_t)BN * HSTR * 2 + 2 * (size_t)XB_BYTES;
    gru_encoder_kernel<<<2, 1024, enc_lds, stream>>>(
        Xbf, WihE, WhhE, bih_e, bhh_e, WihBE, WhhBE, bih_be, bhh_be, hidden);

    // 3) MLP head: mu / logvar (written straight into d_out), z, dec_h
    fc_kernel<<<g1(BN * EE), 256, 0, stream>>>(hidden, W_mu0, b_mu0, t_mu, BN, EE, EE, 1);
    fc_kernel<<<g1(BN * LG), 256, 0, stream>>>(t_mu, W_mu1, b_mu1, out_mu, BN, LG, EE, 0);
    fc_kernel<<<g1(BN * EE), 256, 0, stream>>>(hidden, W_lv0, b_lv0, t_lv, BN, EE, EE, 1);
    fc_kernel<<<g1(BN * LG), 256, 0, stream>>>(t_lv, W_lv1, b_lv1, out_lv, BN, LG, EE, 0);
    zcat_kernel<<<g1(BN * LL), 256, 0, stream>>>(out_mu, out_lv, eps, genre, zc);
    fc_kernel<<<g1(BN * LL), 256, 0, stream>>>(zc, W_lat0, b_lat0, t_lat, BN, LL, LL, 1);
    fc_kernel<<<g1(BN * HN), 256, 0, stream>>>(t_lat, W_lat1, b_lat1, dech, BN, HN, LL, 0);

    // 4) decoder recurrence with output feedback (persistent WMMA workgroup)
    const size_t dec_lds = (size_t)BN * HN * 2 + (size_t)BN * INN * 2;
    gru_decoder_kernel<<<1, 1024, dec_lds, stream>>>(
        x, dech, WihD, WhhD, bih_d, bhh_d, WdB, b_dec1, out_notes);
}
